// Attention_62139586838759
// MI455X (gfx1250) — matrix operand order repaired and checked by reading, not compiled, not run on hardware
//
#include <hip/hip_runtime.h>
#include <hip/hip_bf16.h>
#include <math.h>

typedef __attribute__((ext_vector_type(16))) __bf16 v16bf;
typedef __attribute__((ext_vector_type(8)))  __bf16 v8bf;
typedef __attribute__((ext_vector_type(4)))  __bf16 v4bf;
typedef __attribute__((ext_vector_type(8)))  float  v8f;
typedef __attribute__((ext_vector_type(4)))  float  v4f;

// Problem sizes (fixed by the reference)
constexpr int BSZ  = 128;
constexpr int NCTX = 2048;
constexpr int DK   = 512;   // D
constexpr int HD   = 512;   // H
constexpr int ROWS = 64;    // rows of (b,n) per workgroup; 64 | 2048

// ---------------------------------------------------------------------------
// Kernel 0a: convert Wc (H x D fp32) to bf16 (kept in ws, L2-resident 512KB)
// ---------------------------------------------------------------------------
__global__ __launch_bounds__(256)
void wc_to_bf16_kernel(const float* __restrict__ Wc, __bf16* __restrict__ wcb) {
  int idx = blockIdx.x * 256 + threadIdx.x;           // HD*DK / 256 blocks
  wcb[idx] = (__bf16)Wc[idx];
}

// ---------------------------------------------------------------------------
// Kernel 0b: inp = x @ Wi^T + bi   (tiny: 128x512x512, VALU is fine)
// ---------------------------------------------------------------------------
__global__ __launch_bounds__(256)
void input_linear_kernel(const float* __restrict__ x, const float* __restrict__ Wi,
                         const float* __restrict__ bi, float* __restrict__ inp) {
  int idx = blockIdx.x * 256 + threadIdx.x;           // over B*H
  int b = idx >> 9;
  int h = idx & (HD - 1);
  const v4f* xa = (const v4f*)(x + (size_t)b * DK);
  const v4f* wa = (const v4f*)(Wi + (size_t)h * DK);
  float acc = 0.f;
  #pragma unroll 4
  for (int i = 0; i < DK / 4; ++i) {
    v4f xv = xa[i], wv = wa[i];
    acc += xv[0]*wv[0] + xv[1]*wv[1] + xv[2]*wv[2] + xv[3]*wv[3];
  }
  inp[idx] = acc + bi[h];
}

// ---------------------------------------------------------------------------
// Kernel 1: fused ctx GEMM (WMMA bf16 -> f32) + bias + V.tanh(inp+ctx) score
//   - each WG: 64 rows of (b,n) x all 512 h columns, K = 512
//   - A (context rows) staged in LDS as bf16 (64KB)
//   - each wave holds a 64x64 register macro-tile: 4 row tiles x 4 col tiles
//     (16 live f32 accumulators = 128 VGPRs). The k-loop is the ONLY rolled
//     loop; every A/B fragment loaded in an iteration is consumed in that
//     iteration (4x reuse each) -> nothing for LICM to hoist, no spills.
//   - epilogue: store ctx tile (fp32), reduce V[h]*tanh(inp+ctx) over h
// ---------------------------------------------------------------------------
__global__ __launch_bounds__(256)
void ctx_gemm_score_kernel(const float* __restrict__ context,
                           const __bf16* __restrict__ wcb,
                           const float* __restrict__ bc,
                           const float* __restrict__ inp,
                           const float* __restrict__ V,
                           float* __restrict__ ctx_out,
                           float* __restrict__ att_out) {
  __shared__ __bf16 ldsA[ROWS * DK];   // 64 KB
  __shared__ float  attAcc[ROWS];

  const int tid      = threadIdx.x;
  const int row_base = blockIdx.x * ROWS;        // multiple of 64; 64 | 2048
  const int b        = row_base / NCTX;          // constant across the block

  if (tid < ROWS) attAcc[tid] = 0.f;

  // Stage 64 x 512 fp32 context rows -> bf16 LDS (float4 loads, v4bf stores)
  const float* src = context + (size_t)row_base * DK;
  #pragma unroll
  for (int i = 0; i < (ROWS * DK) / (256 * 4); ++i) {   // 32 iterations
    int e = (i * 256 + tid) * 4;
    v4f cv = *(const v4f*)(src + e);
    v4bf bv;
    bv[0] = (__bf16)cv[0]; bv[1] = (__bf16)cv[1];
    bv[2] = (__bf16)cv[2]; bv[3] = (__bf16)cv[3];
    *(v4bf*)(&ldsA[e]) = bv;
  }
  __syncthreads();

  const int w    = tid >> 5;        // wave 0..7
  const int lane = tid & 31;
  const int half = lane >> 4;       // 0: lanes 0-15, 1: lanes 16-31
  const int l15  = lane & 15;

  // A fragment base per row tile rt: row M = rt*16 + l15; per ISA 16-bit A
  // layout the 16 elements are two contiguous 8-element runs:
  //   k0 + half*8 .. +7   and   k0 + 16 + half*8 .. +7
  const __bf16* abase = &ldsA[l15 * DK + half * 8];

  // B fragment: wave w owns col tiles ct = w*4 .. w*4+3; lane's column is
  // hcol = w*64 + it*16 + l15, K runs = k0 + half*8 .. +7 and k0 + 16 + half*8 .. +7
  const __bf16* bbase = wcb + (size_t)(w * 64 + l15) * DK + half * 8;

  v8f acc[4][4] = {};   // [col tile it][row tile rt]

  #pragma unroll 1      // keep rolled: all operands consumed per iteration
  for (int k0 = 0; k0 < DK; k0 += 32) {
    v16bf a[4];
    #pragma unroll
    for (int rt = 0; rt < 4; ++rt) {
      ((v8bf*)&a[rt])[0] = *(const v8bf*)(abase + rt * 16 * DK + k0);
      ((v8bf*)&a[rt])[1] = *(const v8bf*)(abase + rt * 16 * DK + k0 + 16);
    }
    #pragma unroll
    for (int it = 0; it < 4; ++it) {
      v16bf bm;
      ((v8bf*)&bm)[0] = *(const v8bf*)(bbase + (size_t)it * 16 * DK + k0);
      ((v8bf*)&bm)[1] = *(const v8bf*)(bbase + (size_t)it * 16 * DK + k0 + 16);
      #pragma unroll
      for (int rt = 0; rt < 4; ++rt) {
        acc[it][rt] = __builtin_amdgcn_wmma_f32_16x16x32_bf16(
            false, a[rt], false, bm, (short)0, acc[it][rt], false, false);
      }
    }
  }

  // Epilogue: C/D layout -> VGPR j holds (M = j + 8*half, N = l15)
  #pragma unroll
  for (int it = 0; it < 4; ++it) {
    const int hcol = w * 64 + it * 16 + l15;
    const float bcv  = bc[hcol];
    const float Vv   = V[hcol];
    const float inpv = inp[(size_t)b * HD + hcol];

    #pragma unroll
    for (int rt = 0; rt < 4; ++rt) {
      float* cbase = ctx_out +
          ((size_t)(row_base + rt * 16 + half * 8)) * HD + hcol;
      float s[8];
      #pragma unroll
      for (int j = 0; j < 8; ++j) {
        float cv = acc[it][rt][j] + bcv;
        cbase[(size_t)j * HD] = cv;            // store ctx (fp32)
        s[j] = Vv * tanhf(inpv + cv);          // score contribution
      }
      // Reduce over h within each 16-lane half, accumulate per-row in LDS
      #pragma unroll
      for (int j = 0; j < 8; ++j) {
        float v = s[j];
        v += __shfl_xor(v, 8, 16);
        v += __shfl_xor(v, 4, 16);
        v += __shfl_xor(v, 2, 16);
        v += __shfl_xor(v, 1, 16);
        if (l15 == 0) atomicAdd(&attAcc[rt * 16 + half * 8 + j], v);  // ds_add_f32
      }
    }
  }

  __syncthreads();
  if (tid < ROWS) att_out[row_base + tid] = attAcc[tid];
}

// ---------------------------------------------------------------------------
// Kernel 2: masked softmax over N per batch row -> alpha, log_p
// ---------------------------------------------------------------------------
__global__ __launch_bounds__(256)
void softmax_kernel(const float* __restrict__ att, const int* __restrict__ mask,
                    float* __restrict__ alpha, float* __restrict__ logp) {
  __shared__ float red[256];
  const int b = blockIdx.x, tid = threadIdx.x;
  float vals[NCTX / 256];
  float mx = -INFINITY;
  #pragma unroll
  for (int i = 0; i < NCTX / 256; ++i) {
    int n = tid + i * 256;
    float a = att[(size_t)b * NCTX + n];
    if (mask[(size_t)b * NCTX + n] != 0) a = -INFINITY;
    vals[i] = a;
    mx = fmaxf(mx, a);
  }
  red[tid] = mx; __syncthreads();
  for (int s = 128; s > 0; s >>= 1) {
    if (tid < s) red[tid] = fmaxf(red[tid], red[tid + s]);
    __syncthreads();
  }
  mx = red[0]; __syncthreads();

  float sum = 0.f;
  #pragma unroll
  for (int i = 0; i < NCTX / 256; ++i) sum += expf(vals[i] - mx);
  red[tid] = sum; __syncthreads();
  for (int s = 128; s > 0; s >>= 1) {
    if (tid < s) red[tid] += red[tid + s];
    __syncthreads();
  }
  sum = red[0];

  const float inv = 1.f / sum;
  const float lse = logf(sum);
  #pragma unroll
  for (int i = 0; i < NCTX / 256; ++i) {
    int n = tid + i * 256;
    float sh = vals[i] - mx;
    alpha[(size_t)b * NCTX + n] = expf(sh) * inv;
    logp[(size_t)b * NCTX + n]  = sh - lse;
  }
}

// ---------------------------------------------------------------------------
// Kernel 3: hidden[b,h] = sum_n alpha[b,n] * ctx[b,n,h]  (coalesced over h)
// ---------------------------------------------------------------------------
__global__ __launch_bounds__(256)
void mix_kernel(const float* __restrict__ ctx, const float* __restrict__ alpha,
                float* __restrict__ hidden) {
  __shared__ float la[NCTX];   // 8 KB
  const int b = blockIdx.x;
  const int h = blockIdx.y * 256 + threadIdx.x;
  for (int i = threadIdx.x; i < NCTX; i += 256) la[i] = alpha[(size_t)b * NCTX + i];
  __syncthreads();
  const float* cb = ctx + (size_t)b * NCTX * HD + h;
  float acc = 0.f;
  #pragma unroll 4
  for (int n = 0; n < NCTX; ++n) acc += la[n] * cb[(size_t)n * HD];
  hidden[(size_t)b * HD + h] = acc;
}

// ---------------------------------------------------------------------------
// Host-side launcher
//   inputs (setup_inputs order): x, context, mask, Wi, bi, Wc, bc, V
//   d_out = [hidden (B*H)] [alpha (B*N)] [log_p (B*N)]  (fp32)
//   d_ws  = [ctx fp32 512MB][att 1MB][inp 256KB][Wc bf16 512KB]
// ---------------------------------------------------------------------------
extern "C" void kernel_launch(void* const* d_in, const int* in_sizes, int n_in,
                              void* d_out, int out_size, void* d_ws, size_t ws_size,
                              hipStream_t stream) {
  const float* x       = (const float*)d_in[0];
  const float* context = (const float*)d_in[1];
  const int*   mask    = (const int*)  d_in[2];
  const float* Wi      = (const float*)d_in[3];
  const float* bi      = (const float*)d_in[4];
  const float* Wc      = (const float*)d_in[5];
  const float* bc      = (const float*)d_in[6];
  const float* V       = (const float*)d_in[7];

  float* out_hidden = (float*)d_out;
  float* out_alpha  = out_hidden + (size_t)BSZ * HD;
  float* out_logp   = out_alpha  + (size_t)BSZ * NCTX;

  const size_t CTX_BYTES = (size_t)BSZ * NCTX * HD * sizeof(float);
  float*  ws_ctx = (float*)d_ws;
  float*  ws_att = (float*)((char*)d_ws + CTX_BYTES);
  float*  ws_inp = ws_att + (size_t)BSZ * NCTX;
  __bf16* ws_wcb = (__bf16*)(ws_inp + (size_t)BSZ * HD);

  // 0a: Wc -> bf16
  wc_to_bf16_kernel<<<(HD * DK) / 256, 256, 0, stream>>>(Wc, ws_wcb);
  // 0b: inp = x @ Wi^T + bi
  input_linear_kernel<<<(BSZ * HD) / 256, 256, 0, stream>>>(x, Wi, bi, ws_inp);
  // 1: fused ctx GEMM (WMMA) + score
  ctx_gemm_score_kernel<<<(BSZ * NCTX) / ROWS, 256, 0, stream>>>(
      context, ws_wcb, bc, ws_inp, V, ws_ctx, ws_att);
  // 2: masked softmax -> alpha, log_p
  softmax_kernel<<<BSZ, 256, 0, stream>>>(ws_att, mask, out_alpha, out_logp);
  // 3: hidden = ctx^T @ alpha
  mix_kernel<<<dim3(BSZ, HD / 256), 256, 0, stream>>>(ws_ctx, out_alpha, out_hidden);
}
